// FeedForward_61598420959519
// MI455X (gfx1250) — matrix operand order repaired and checked by reading, not compiled, not run on hardware
//
#include <hip/hip_runtime.h>
#include <math.h>

// ---------------------------------------------------------------------------
// Channel attention block, algebraically folded:
//   Wq' = Wq*Wsplit (likewise k,v)  -> QKV = Wall(576x192) * x_in(192x16384)
//   logits = (K Q^T) * rnk[c]*rnq[d]; softmax over d
//   Mb = (Wout*Wproj) * attn        -> out = Mb * V + (Wout*b_proj)
// Big GEMMs: v_wmma_f32_16x16x32_bf16 (wave32). S-GEMM uses double-buffered
// async global->LDS copies (ASYNCcnt) when the toolchain exposes them.
// ---------------------------------------------------------------------------

typedef __bf16 bf16_t;
typedef __attribute__((ext_vector_type(16))) __bf16 v16bf;
typedef __attribute__((ext_vector_type(8)))  float  v8f;

#define CCH   192        // channels
#define NSP   16384      // spatial h*w
#define NB    8          // batch
#define LDP   (CCH + 8)  // LDS pitch for K=192 tiles (400 B rows, 16B aligned)
#define SLP   (64 + 8)   // LDS pitch for 64-col S-GEMM tiles (144 B rows)

#if defined(__HIP_DEVICE_COMPILE__) && __has_builtin(__builtin_amdgcn_global_load_async_to_lds_b128)
#define HAVE_ASYNC 1
#else
#define HAVE_ASYNC 0
#endif

#if HAVE_ASYNC
// Exact parameter type reported by the toolchain for the async-LDS builtin:
// '__attribute__((__vector_size__(4 * sizeof(int)))) int *'
typedef int v4i_async __attribute__((vector_size(4 * sizeof(int))));
#endif

__device__ __forceinline__ void wait_async_cnt0() {
#if HAVE_ASYNC
#if __has_builtin(__builtin_amdgcn_s_wait_asynccnt)
  __builtin_amdgcn_s_wait_asynccnt(0);
#else
  asm volatile("s_wait_asynccnt 0x0" ::: "memory");
#endif
#endif
}

// ---- WMMA fragment gathers (wave32, 16x16x32 bf16) -------------------------
// A (16x32, M x K): lanes 0-15 M=0-15; VGPR0-3: K = half*8 + 0..7,
//                   VGPR4-7: K = 16 + half*8 + 0..7
__device__ __forceinline__ v16bf load_a_frag(const bf16_t* lds, int lda,
                                             int m0, int k0, int lane16, int half) {
  const bf16_t* row = lds + (m0 + lane16) * lda + k0 + half * 8;
  v16bf a;
#pragma unroll
  for (int e = 0; e < 8; ++e)  a[e] = row[e];
#pragma unroll
  for (int e = 8; e < 16; ++e) a[e] = row[e + 8];   // +16 block
  return a;
}

// B (32x16, K x N): lanes 0-15 (N) hold K=half*8 + 0..7 and 16 + half*8 + 0..7
// (same order as A), operand stored as [N-row][K].
__device__ __forceinline__ v16bf load_b_frag(const bf16_t* lds, int ldb,
                                             int n0, int k0, int lane16, int half) {
  const bf16_t* row = lds + (n0 + lane16) * ldb + k0 + half * 8;
  v16bf b;
#pragma unroll
  for (int e = 0; e < 8; ++e)  b[e] = row[e];
#pragma unroll
  for (int e = 8; e < 16; ++e) b[e] = row[e + 8];
  return b;
}

#define WMMA_BF16(acc, a, b) \
  acc = __builtin_amdgcn_wmma_f32_16x16x32_bf16(false, a, false, b, (short)0, acc, false, false)

// ---- tiny weight-combine kernels ------------------------------------------
__global__ __launch_bounds__(CCH) void combine_qkv_w(
    const float* __restrict__ w_split, const float* __restrict__ w_q,
    const float* __restrict__ w_k, const float* __restrict__ w_v,
    bf16_t* __restrict__ Wall) {
  int m = blockIdx.x;            // 0..575 (uniform scalar divide below)
  int k = threadIdx.x;           // 0..191
  int sel = m / CCH, c = m - sel * CCH;
  const float* wsel = (sel == 0) ? w_q : (sel == 1) ? w_k : w_v;
  float s = 0.f;
  for (int j = 0; j < CCH; ++j) s += wsel[c * CCH + j] * w_split[j * CCH + k];
  Wall[(size_t)m * CCH + k] = (bf16_t)s;
}

__global__ __launch_bounds__(CCH) void combine_out_w(
    const float* __restrict__ w_proj, const float* __restrict__ b_proj,
    const float* __restrict__ w_out, float* __restrict__ Wpo,
    float* __restrict__ bpo) {
  int o = blockIdx.x, c = threadIdx.x;
  float s = 0.f;
  for (int j = 0; j < CCH; ++j) s += w_out[o * CCH + j] * w_proj[j * CCH + c];
  Wpo[o * CCH + c] = s;
  if (c == 0) {
    float sb = 0.f;
    for (int j = 0; j < CCH; ++j) sb += w_out[o * CCH + j] * b_proj[j];
    bpo[o] = sb;
  }
}

// ---- QKV GEMM: Wall(576x192 bf16) * x_in[b](192x16384 f32) -> QKV bf16 -----
__global__ __launch_bounds__(128) void qkv_gemm(
    const float* __restrict__ x_in, const bf16_t* __restrict__ Wall,
    bf16_t* __restrict__ QKV) {
  __shared__ bf16_t ldsA[64 * LDP];   // [m][k]
  __shared__ bf16_t ldsB[64 * LDP];   // [n][k] (transposed fill)
  int tid = threadIdx.x;
  int n0 = blockIdx.x * 64;
  int m0 = blockIdx.y * 64;           // 9 blocks -> 576 rows
  int sel = blockIdx.y / 3;           // uniform: which of q/k/v
  int c_base = m0 - sel * CCH;        // channel offset within that matrix
  int b = blockIdx.z;

  // A fill: 64 rows x 192 bf16 = 24 uint4 per row; 2 threads per row.
  {
    int r = tid >> 1, s0 = (tid & 1) * 12;
    const uint4* srow = (const uint4*)(Wall + (size_t)(m0 + r) * CCH);
    uint4* drow = (uint4*)(ldsA + r * LDP);
#pragma unroll
    for (int s = 0; s < 12; ++s) drow[s0 + s] = srow[s0 + s];
  }
  // B fill (transpose f32 -> bf16): row k has 64 floats = 16 float4.
  const float* xb = x_in + (size_t)b * CCH * NSP + n0;
  for (int i = tid; i < CCH * 16; i += 128) {
    int k = i >> 4, s = i & 15;
    float4 v = ((const float4*)(xb + (size_t)k * NSP))[s];
    int n = s * 4;
    ldsB[(n + 0) * LDP + k] = (bf16_t)v.x;
    ldsB[(n + 1) * LDP + k] = (bf16_t)v.y;
    ldsB[(n + 2) * LDP + k] = (bf16_t)v.z;
    ldsB[(n + 3) * LDP + k] = (bf16_t)v.w;
  }
  __syncthreads();

  int wid = tid >> 5, lane = tid & 31, half = lane >> 4, lane16 = lane & 15;
  int wm = (wid >> 1) * 32, wn = (wid & 1) * 32;
  v8f acc[2][2] = {};
#pragma unroll
  for (int kc = 0; kc < 6; ++kc) {
    int k0 = kc * 32;
    v16bf af[2], bfv[2];
    af[0]  = load_a_frag(ldsA, LDP, wm,      k0, lane16, half);
    af[1]  = load_a_frag(ldsA, LDP, wm + 16, k0, lane16, half);
    bfv[0] = load_b_frag(ldsB, LDP, wn,      k0, lane16, half);
    bfv[1] = load_b_frag(ldsB, LDP, wn + 16, k0, lane16, half);
#pragma unroll
    for (int i = 0; i < 2; ++i)
#pragma unroll
      for (int j = 0; j < 2; ++j) WMMA_BF16(acc[i][j], af[i], bfv[j]);
  }
  bf16_t* dstm = QKV + ((size_t)sel * NB + b) * CCH * NSP;
#pragma unroll
  for (int i = 0; i < 2; ++i)
#pragma unroll
    for (int j = 0; j < 2; ++j) {
      int cb = c_base + wm + i * 16 + 8 * half;
      int nc = n0 + wn + j * 16 + lane16;
#pragma unroll
      for (int r = 0; r < 8; ++r)
        dstm[(size_t)(cb + r) * NSP + nc] = (bf16_t)acc[i][j][r];
    }
}

// ---- row L2-norm reciprocals for Q (sel 0) and K (sel 1) -------------------
__global__ __launch_bounds__(256) void rownorm(const bf16_t* __restrict__ QKV,
                                               float* __restrict__ rn) {
  int idx = blockIdx.x;                        // 0..2*NB*CCH-1 (Q rows, K rows)
  const bf16_t* row = QKV + (size_t)idx * NSP;
  float s = 0.f;
  for (int i = threadIdx.x; i < NSP; i += 256) {
    float v = (float)row[i];
    s += v * v;
  }
  __shared__ float red[256];
  red[threadIdx.x] = s;
  __syncthreads();
  for (int off = 128; off > 0; off >>= 1) {
    if (threadIdx.x < off) red[threadIdx.x] += red[threadIdx.x + off];
    __syncthreads();
  }
  if (threadIdx.x == 0) rn[idx] = 1.f / fmaxf(sqrtf(red[0]), 1e-12f);
}

// ---- S = K * Q^T, split-K, double-buffered (async) LDS staging -------------
__device__ __forceinline__ void fill_tile64(bf16_t* dst, const bf16_t* src, int tid) {
  // 64 rows x 64 bf16 cols = 512 x 16B moves; src row stride NSP, dst pitch SLP
#pragma unroll
  for (int t = 0; t < 4; ++t) {
    int i = tid + t * 128;
    int r = i >> 3, s = i & 7;
#if HAVE_ASYNC
    __builtin_amdgcn_global_load_async_to_lds_b128(
        (v4i_async*)(src + (size_t)r * NSP + s * 8),
        (v4i_async*)(dst + r * SLP + s * 8), 0, 0);
#else
    *(uint4*)(dst + r * SLP + s * 8) =
        *(const uint4*)(src + (size_t)r * NSP + s * 8);
#endif
  }
}

__global__ __launch_bounds__(128) void s_gemm(const bf16_t* __restrict__ Kmat,
                                              const bf16_t* __restrict__ Qmat,
                                              float* __restrict__ Spart) {
  __shared__ bf16_t ldsK[2][64 * SLP];  // [c][n]
  __shared__ bf16_t ldsQ[2][64 * SLP];  // [d][n]  (B operand already [N][K])
  int tid = threadIdx.x;
  int region = blockIdx.x;              // 0..8 -> 3x3 of 64x64
  int slice = blockIdx.y, b = blockIdx.z;
  int c0 = (region / 3) * 64, d0 = (region % 3) * 64;   // uniform scalar
  const bf16_t* Kt = Kmat + ((size_t)b * CCH + c0) * NSP + slice * 2048;
  const bf16_t* Qt = Qmat + ((size_t)b * CCH + d0) * NSP + slice * 2048;
  int wid = tid >> 5, lane = tid & 31, half = lane >> 4, lane16 = lane & 15;
  int wm = (wid >> 1) * 32, wn = (wid & 1) * 32;
  v8f acc[2][2] = {};

  fill_tile64(ldsK[0], Kt, tid);
  fill_tile64(ldsQ[0], Qt, tid);
  wait_async_cnt0();
  __syncthreads();

  for (int blk = 0; blk < 32; ++blk) {
    int cur = blk & 1;
    if (blk + 1 < 32) {      // prefetch next chunk into the other buffer
      fill_tile64(ldsK[cur ^ 1], Kt + (blk + 1) * 64, tid);
      fill_tile64(ldsQ[cur ^ 1], Qt + (blk + 1) * 64, tid);
    }
    const bf16_t* bK = ldsK[cur];
    const bf16_t* bQ = ldsQ[cur];
#pragma unroll
    for (int kc = 0; kc < 2; ++kc) {
      int k0 = kc * 32;
      v16bf af[2], bfv[2];
      af[0]  = load_a_frag(bK, SLP, wm,      k0, lane16, half);
      af[1]  = load_a_frag(bK, SLP, wm + 16, k0, lane16, half);
      bfv[0] = load_b_frag(bQ, SLP, wn,      k0, lane16, half);
      bfv[1] = load_b_frag(bQ, SLP, wn + 16, k0, lane16, half);
#pragma unroll
      for (int i = 0; i < 2; ++i)
#pragma unroll
        for (int j = 0; j < 2; ++j) WMMA_BF16(acc[i][j], af[i], bfv[j]);
    }
    wait_async_cnt0();
    __syncthreads();
  }

  float* Sp = Spart + ((size_t)b * 8 + slice) * CCH * CCH;
#pragma unroll
  for (int i = 0; i < 2; ++i)
#pragma unroll
    for (int j = 0; j < 2; ++j) {
      int cb = c0 + wm + i * 16 + 8 * half;
      int dc = d0 + wn + j * 16 + lane16;
#pragma unroll
      for (int r = 0; r < 8; ++r) Sp[(size_t)(cb + r) * CCH + dc] = acc[i][j][r];
    }
}

// ---- softmax over d with fused norm scaling --------------------------------
__global__ __launch_bounds__(256) void softmax_attn(const float* __restrict__ Spart,
                                                    const float* __restrict__ rn,
                                                    float* __restrict__ attn) {
  int b = blockIdx.x / CCH, c = blockIdx.x - b * CCH;
  int d = threadIdx.x;
  const float* rnq = rn;
  const float* rnk = rn + NB * CCH;
  float logit = -INFINITY;
  if (d < CCH) {
    float s = 0.f;
    for (int sl = 0; sl < 8; ++sl)
      s += Spart[(((size_t)b * 8 + sl) * CCH + c) * CCH + d];
    logit = s * rnk[b * CCH + c] * rnq[b * CCH + d];
  }
  __shared__ float red[256];
  red[d] = logit;
  __syncthreads();
  for (int off = 128; off > 0; off >>= 1) {
    if (d < off) red[d] = fmaxf(red[d], red[d + off]);
    __syncthreads();
  }
  float mx = red[0];
  __syncthreads();
  float e = (d < CCH) ? __expf(logit - mx) : 0.f;
  red[d] = e;
  __syncthreads();
  for (int off = 128; off > 0; off >>= 1) {
    if (d < off) red[d] += red[d + off];
    __syncthreads();
  }
  if (d < CCH) attn[((size_t)b * CCH + c) * CCH + d] = e / red[0];
}

// ---- Mb = Wpo * attn[b]  (192x192 per batch, tiny) -------------------------
__global__ __launch_bounds__(CCH) void make_mb(const float* __restrict__ Wpo,
                                               const float* __restrict__ attn,
                                               bf16_t* __restrict__ Mb) {
  int o = blockIdx.x, b = blockIdx.y, d = threadIdx.x;
  const float* ab = attn + (size_t)b * CCH * CCH;
  float s = 0.f;
  for (int c = 0; c < CCH; ++c) s += Wpo[o * CCH + c] * ab[c * CCH + d];
  Mb[((size_t)b * CCH + o) * CCH + d] = (bf16_t)s;
}

// ---- OUT = Mb[b] * V[b] + bpo, f32 to d_out --------------------------------
__global__ __launch_bounds__(128) void final_gemm(const bf16_t* __restrict__ Vmat,
                                                  const bf16_t* __restrict__ Mb,
                                                  const float* __restrict__ bpo,
                                                  float* __restrict__ out) {
  __shared__ bf16_t ldsA[64 * LDP];   // [o][d]
  __shared__ bf16_t ldsB[64 * LDP];   // [n][d]
  int tid = threadIdx.x;
  int n0 = blockIdx.x * 64;
  int m0 = blockIdx.y * 64;           // 3 blocks -> 192 rows
  int b  = blockIdx.z;
  const bf16_t* Vb = Vmat + (size_t)b * CCH * NSP + n0;

  // A fill: 64 rows x 24 uint4, 2 threads per row.
  {
    int r = tid >> 1, s0 = (tid & 1) * 12;
    const uint4* srow = (const uint4*)(Mb + ((size_t)b * CCH + m0 + r) * CCH);
    uint4* drow = (uint4*)(ldsA + r * LDP);
#pragma unroll
    for (int s = 0; s < 12; ++s) drow[s0 + s] = srow[s0 + s];
  }
  // B fill (bf16 transpose): row k has 64 bf16 = 8 uint4.
  for (int i = tid; i < CCH * 8; i += 128) {
    int k = i >> 3, s = i & 7;
    union { uint4 u; bf16_t h[8]; } w;
    w.u = ((const uint4*)(Vb + (size_t)k * NSP))[s];
    int n = s * 8;
#pragma unroll
    for (int t = 0; t < 8; ++t) ldsB[(n + t) * LDP + k] = w.h[t];
  }
  __syncthreads();

  int wid = tid >> 5, lane = tid & 31, half = lane >> 4, lane16 = lane & 15;
  int wm = (wid >> 1) * 32, wn = (wid & 1) * 32;
  v8f acc[2][2] = {};
#pragma unroll
  for (int kc = 0; kc < 6; ++kc) {
    int k0 = kc * 32;
    v16bf af[2], bfv[2];
    af[0]  = load_a_frag(ldsA, LDP, wm,      k0, lane16, half);
    af[1]  = load_a_frag(ldsA, LDP, wm + 16, k0, lane16, half);
    bfv[0] = load_b_frag(ldsB, LDP, wn,      k0, lane16, half);
    bfv[1] = load_b_frag(ldsB, LDP, wn + 16, k0, lane16, half);
#pragma unroll
    for (int i = 0; i < 2; ++i)
#pragma unroll
      for (int j = 0; j < 2; ++j) WMMA_BF16(acc[i][j], af[i], bfv[j]);
  }
#pragma unroll
  for (int i = 0; i < 2; ++i)
#pragma unroll
    for (int j = 0; j < 2; ++j) {
      int mb = m0 + wm + i * 16 + 8 * half;
      int nc = n0 + wn + j * 16 + lane16;
#pragma unroll
      for (int r = 0; r < 8; ++r) {
        int o = mb + r;
        out[((size_t)b * CCH + o) * NSP + nc] = acc[i][j][r] + bpo[o];
      }
    }
}

// ---------------------------------------------------------------------------
extern "C" void kernel_launch(void* const* d_in, const int* in_sizes, int n_in,
                              void* d_out, int out_size, void* d_ws, size_t ws_size,
                              hipStream_t stream) {
  const float* x_in    = (const float*)d_in[0];
  const float* w_split = (const float*)d_in[1];
  const float* w_q     = (const float*)d_in[2];
  const float* w_k     = (const float*)d_in[3];
  const float* w_v     = (const float*)d_in[4];
  const float* w_proj  = (const float*)d_in[5];
  const float* b_proj  = (const float*)d_in[6];
  const float* w_out   = (const float*)d_in[7];
  float* out = (float*)d_out;

  // workspace layout (all offsets 256B-aligned by construction; ~155 MB total)
  char* ws = (char*)d_ws;
  size_t off = 0;
  bf16_t* Wall = (bf16_t*)(ws + off); off += (size_t)3 * CCH * CCH * sizeof(bf16_t);
  float*  Wpo  = (float*) (ws + off); off += (size_t)CCH * CCH * sizeof(float);
  float*  bpo  = (float*) (ws + off); off += 1024;
  bf16_t* QKV  = (bf16_t*)(ws + off); off += (size_t)3 * NB * CCH * NSP * sizeof(bf16_t);
  float*  rn   = (float*) (ws + off); off += (size_t)2 * NB * CCH * sizeof(float);
  float*  Spart= (float*) (ws + off); off += (size_t)NB * 8 * CCH * CCH * sizeof(float);
  float*  attn = (float*) (ws + off); off += (size_t)NB * CCH * CCH * sizeof(float);
  bf16_t* Mb   = (bf16_t*)(ws + off); off += (size_t)NB * CCH * CCH * sizeof(bf16_t);
  (void)off; (void)ws_size; (void)in_sizes; (void)n_in; (void)out_size;

  const bf16_t* Qmat = QKV;                                 // [NB][CCH][NSP]
  const bf16_t* Kmat = QKV + (size_t)NB * CCH * NSP;
  const bf16_t* Vmat = QKV + (size_t)2 * NB * CCH * NSP;

  combine_qkv_w<<<dim3(3 * CCH), dim3(CCH), 0, stream>>>(w_split, w_q, w_k, w_v, Wall);
  combine_out_w<<<dim3(CCH), dim3(CCH), 0, stream>>>(w_proj, b_proj, w_out, Wpo, bpo);
  qkv_gemm<<<dim3(NSP / 64, 9, NB), dim3(128), 0, stream>>>(x_in, Wall, QKV);
  rownorm<<<dim3(2 * NB * CCH), dim3(256), 0, stream>>>(QKV, rn);
  s_gemm<<<dim3(9, 8, NB), dim3(128), 0, stream>>>(Kmat, Qmat, Spart);
  softmax_attn<<<dim3(NB * CCH), dim3(256), 0, stream>>>(Spart, rn, attn);
  make_mb<<<dim3(CCH, NB), dim3(CCH), 0, stream>>>(Wpo, attn, Mb);
  final_gemm<<<dim3(NSP / 64, 3, NB), dim3(128), 0, stream>>>(Vmat, Mb, bpo, out);
}
